// KNF_Ensemble_63264868270723
// MI455X (gfx1250) — matrix operand order repaired and checked by reading, not compiled, not run on hardware
//
#include <hip/hip_runtime.h>
#include <math.h>

// ---------------------------------------------------------------------------
// Types for CDNA5 WMMA (gfx1250, wave32)
// ---------------------------------------------------------------------------
typedef __attribute__((ext_vector_type(16))) __bf16 bf16x16;
typedef __attribute__((ext_vector_type(8)))  float  v8f;
typedef __attribute__((ext_vector_type(8)))  unsigned short ushort8;

union Frag16 { bf16x16 v; ushort8 h[2]; unsigned short s[16]; };

__device__ __forceinline__ unsigned short f32_to_bf16_bits(float f) {
  unsigned int u = __float_as_uint(f);
  u += 0x7FFFu + ((u >> 16) & 1u);            // round-to-nearest-even
  return (unsigned short)(u >> 16);
}

__device__ __forceinline__ float gelu_f(float x) {
  return 0.5f * x * (1.0f + erff(x * 0.70710678118654752440f));
}

__device__ __forceinline__ void atomicMaxF(float* addr, float val) {
  int* a = (int*)addr;
  int cur = *a;
  while (__int_as_float(cur) < val) {
    int prev = atomicCAS(a, cur, __float_as_int(val));
    if (prev == cur) break;
    cur = prev;
  }
}

// ---------------------------------------------------------------------------
// Activation prepack: A[M,K] f32 -> Ap[M,Kpad] bf16 bits, zero padded
// ---------------------------------------------------------------------------
__global__ void apack_kernel(const float* __restrict__ A,
                             unsigned short* __restrict__ Ap,
                             int K, int Kpad, int M) {
  long idx = blockIdx.x * (long)blockDim.x + threadIdx.x;
  if (idx >= (long)M * Kpad) return;
  int i = (int)(idx / Kpad), k = (int)(idx - (long)i * Kpad);
  Ap[idx] = (k < K) ? f32_to_bf16_bits(A[(size_t)i * K + k]) : (unsigned short)0;
}

// ---------------------------------------------------------------------------
// Weight prepack: W[K,N] f32 (row-major) -> Wt[N,Kpad] bf16 bits, zero padded
// ---------------------------------------------------------------------------
__global__ void prepack_kernel(const float* __restrict__ W,
                               unsigned short* __restrict__ Wt,
                               int K, int Nn, int Kpad) {
  int idx = blockIdx.x * blockDim.x + threadIdx.x;
  int tot = Nn * Kpad;
  if (idx >= tot) return;
  int n = idx / Kpad, k = idx - n * Kpad;
  Wt[idx] = (k < K) ? f32_to_bf16_bits(W[(size_t)k * Nn + n]) : (unsigned short)0;
}

// ---------------------------------------------------------------------------
// WMMA GEMM: C[M,N] = act( (ACCUM? C:0) + A @ W + bias )
// Ap: packed bf16 [M,Kpad].  Wt: packed bf16 [N,Kpad].
// One wave -> 16x64 tile (4 WMMAs per K-step, A-fragment reused 4x).
// ACT: 0=none 1=gelu 2=relu
// ---------------------------------------------------------------------------
template<int ACT, int ACCUM>
__global__ __launch_bounds__(128)
void gemm_wmma_kernel(const unsigned short* __restrict__ Ap,
                      const unsigned short* __restrict__ Wt,
                      const float* __restrict__ bias,
                      float* __restrict__ C,
                      int M, int Nn, int Kpad) {
  int wave = threadIdx.x >> 5;
  int lane = threadIdx.x & 31;
  int tileN = blockIdx.x * 64;
  int tileM = (blockIdx.y * 4 + wave) * 16;
  if (tileM >= M) return;                        // wave-uniform

  int lh = lane & 15;
  int hi = lane >> 4;                            // 0 or 1

  int arow = tileM + lh; if (arow >= M) arow = M - 1;
  const unsigned short* aptr = Ap + (size_t)arow * Kpad + (hi ? 8 : 0);

  const unsigned short* bptr[4];
#pragma unroll
  for (int t = 0; t < 4; ++t) {
    int bc = tileN + t * 16 + lh; if (bc >= Nn) bc = Nn - 1;
    bptr[t] = Wt + (size_t)bc * Kpad + (hi ? 8 : 0);
  }

  const v8f vzero = {0.f, 0.f, 0.f, 0.f, 0.f, 0.f, 0.f, 0.f};
  v8f acc[4];
#pragma unroll
  for (int t = 0; t < 4; ++t) acc[t] = vzero;

  for (int k0 = 0; k0 < Kpad; k0 += 32) {
    Frag16 af;
    af.h[0] = *(const ushort8*)(aptr + k0);        // K: k0+kbase .. +7
    af.h[1] = *(const ushort8*)(aptr + k0 + 16);   // K: k0+kbase+16 .. +23
#pragma unroll
    for (int t = 0; t < 4; ++t) {
      Frag16 bf;
      bf.h[0] = *(const ushort8*)(bptr[t] + k0);
      bf.h[1] = *(const ushort8*)(bptr[t] + k0 + 16);
      acc[t] = __builtin_amdgcn_wmma_f32_16x16x32_bf16(false, af.v, false, bf.v,
                                                       (short)0, acc[t], false, false);
    }
  }

#pragma unroll
  for (int t = 0; t < 4; ++t) {
    int col = tileN + t * 16 + lh;
    if (col < Nn) {
      float bval = bias ? bias[col] : 0.0f;
#pragma unroll
      for (int r = 0; r < 8; ++r) {
        int row = tileM + r + (hi ? 8 : 0);
        if (row < M) {
          size_t idx = (size_t)row * Nn + col;
          float v = acc[t][r] + bval;
          if (ACCUM) v += C[idx];
          if (ACT == 1) v = gelu_f(v);
          else if (ACT == 2) v = fmaxf(v, 0.0f);
          C[idx] = v;
        }
      }
    }
  }
}

// ---------------------------------------------------------------------------
// Elementwise / utility kernels
// ---------------------------------------------------------------------------
__global__ void fill_kernel(float* p, float v, size_t n) {
  size_t i = blockIdx.x * (size_t)blockDim.x + threadIdx.x;
  if (i < n) p[i] = v;
}

__global__ void axpy_kernel(float* out, const float* in, float w, int n) {
  int i = blockIdx.x * blockDim.x + threadIdx.x;
  if (i < n) out[i] += w * in[i];
}

template<int ACT>
__global__ void bias_act_kernel(const float* in, const float* bias, float* out,
                                size_t rows, int cols) {
  size_t idx = blockIdx.x * (size_t)blockDim.x + threadIdx.x;
  if (idx >= rows * (size_t)cols) return;
  int c = (int)(idx % (size_t)cols);
  float v = in[idx] + (bias ? bias[c] : 0.0f);
  if (ACT == 1) v = gelu_f(v);
  else if (ACT == 2) v = fmaxf(v, 0.0f);
  out[idx] = v;
}

template<int ACT>
__global__ void head_mean_kernel(const float* in, const float* bias, float* out,
                                 int n, int H, int C) {
  int idx = blockIdx.x * blockDim.x + threadIdx.x;
  if (idx >= n * C) return;
  int i = idx / C, c = idx - i * C;
  float s = 0.f;
  for (int h = 0; h < H; ++h) s += in[(size_t)i * H * C + h * C + c];
  s = s / (float)H + (bias ? bias[c] : 0.0f);
  if (ACT == 1) s = gelu_f(s);
  else if (ACT == 2) s = fmaxf(s, 0.0f);
  out[idx] = s;
}

__global__ void layernorm_kernel(float* X, const float* g, const float* b,
                                 int rows, int cols) {
  int r = blockIdx.x * blockDim.x + threadIdx.x;
  if (r >= rows) return;
  float* x = X + (size_t)r * cols;
  float mu = 0.f;
  for (int j = 0; j < cols; ++j) mu += x[j];
  mu /= (float)cols;
  float var = 0.f;
  for (int j = 0; j < cols; ++j) { float d = x[j] - mu; var += d * d; }
  var /= (float)cols;
  float inv = rsqrtf(var + 1e-5f);
  for (int j = 0; j < cols; ++j) x[j] = (x[j] - mu) * inv * g[j] + b[j];
}

// ---------------------------------------------------------------------------
// Segment / scatter kernels
// ---------------------------------------------------------------------------
__global__ void scatter_add_feat_kernel(const float* X, const int* src, const int* dst,
                                        float* out, int E, int F) {
  long idx = blockIdx.x * (long)blockDim.x + threadIdx.x;
  if (idx >= (long)E * F) return;
  int e = (int)(idx / F), f = (int)(idx - (long)e * F);
  atomicAdd(&out[(size_t)dst[e] * F + f], X[(size_t)src[e] * F + f]);
}

__global__ void scatter_max_feat_kernel(const float* X, const int* src, const int* dst,
                                        float* out, int E, int F) {
  long idx = blockIdx.x * (long)blockDim.x + threadIdx.x;
  if (idx >= (long)E * F) return;
  int e = (int)(idx / F), f = (int)(idx - (long)e * F);
  atomicMaxF(&out[(size_t)dst[e] * F + f], X[(size_t)src[e] * F + f]);
}

__global__ void count_dst_kernel(const int* dst, float* cnt, int E) {
  int e = blockIdx.x * blockDim.x + threadIdx.x;
  if (e < E) atomicAdd(&cnt[dst[e]], 1.0f);
}

__global__ void div_cnt_kernel(float* X, const float* cnt, int n, int F) {
  long idx = blockIdx.x * (long)blockDim.x + threadIdx.x;
  if (idx >= (long)n * F) return;
  int i = (int)(idx / F);
  X[idx] /= fmaxf(cnt[i], 1.0f);
}

__global__ void fixinf_kernel(float* X, size_t n) {
  size_t i = blockIdx.x * (size_t)blockDim.x + threadIdx.x;
  if (i >= n) return;
  float v = X[i];
  if (!isfinite(v)) X[i] = 0.0f;
}

// ---------------------------------------------------------------------------
// GATv2 kernels
// ---------------------------------------------------------------------------
__global__ void gat_alpha_kernel(const float* XL, const float* XR, const float* ea,
                                 const float* loopattr, const float* we, const float* att,
                                 const int* src, const int* dst,
                                 float* alpha, float* amax, int E, int n, int H, int C) {
  int idx = blockIdx.x * blockDim.x + threadIdx.x;
  int tot = (E + n) * H;
  if (idx >= tot) return;
  int e = idx / H, h = idx - e * H;
  int s, d; const float* eap;
  if (e < E) { s = src[e]; d = dst[e]; eap = ea + (size_t)e * 7; }
  else       { s = d = e - E;          eap = loopattr + (size_t)s * 7; }
  int HC = H * C;
  float eaf[7];
#pragma unroll
  for (int f = 0; f < 7; ++f) eaf[f] = eap[f];
  const float* xlp = XL + (size_t)s * HC + h * C;
  const float* xrp = XR + (size_t)d * HC + h * C;
  const float* wep = we + h * C;
  const float* atp = att + h * C;
  float acc = 0.f;
  for (int c = 0; c < C; ++c) {
    float ee = 0.f;
#pragma unroll
    for (int f = 0; f < 7; ++f) ee = fmaf(eaf[f], wep[(size_t)f * HC + c], ee);
    float sv = xlp[c] + xrp[c] + ee;
    sv = (sv > 0.f) ? sv : 0.2f * sv;
    acc = fmaf(sv, atp[c], acc);
  }
  alpha[idx] = acc;
  atomicMaxF(&amax[d * H + h], acc);
}

__global__ void seg_exp_kernel(float* alpha, const float* amax, float* asum,
                               const int* dst, int E, int nloops, int H) {
  int idx = blockIdx.x * blockDim.x + threadIdx.x;
  int tot = (E + nloops) * H;
  if (idx >= tot) return;
  int e = idx / H, h = idx - e * H;
  int d = (e < E) ? dst[e] : (e - E);
  float m = amax[d * H + h];
  if (!isfinite(m)) m = 0.0f;
  float ex = expf(alpha[idx] - m);
  alpha[idx] = ex;
  atomicAdd(&asum[d * H + h], ex);
}

__global__ void gat_scatter_kernel(const float* XL, const float* alpha, const float* asum,
                                   const int* src, const int* dst, float* out,
                                   int E, int n, int H, int C) {
  int e = blockIdx.x;
  int t = blockIdx.y * blockDim.x + threadIdx.x;
  int HC = H * C;
  if (t >= HC) return;
  int s, d;
  if (e < E) { s = src[e]; d = dst[e]; } else { s = d = e - E; }
  int h = t / C;
  float a = alpha[(size_t)e * H + h] / (asum[d * H + h] + 1e-16f);
  atomicAdd(&out[(size_t)d * HC + t], XL[(size_t)s * HC + t] * a);
}

// ---------------------------------------------------------------------------
// TransformerConv kernels
// ---------------------------------------------------------------------------
__global__ void tr_alpha_kernel(const float* Q, const float* Km, const float* ea,
                                const float* we, const float* be,
                                const int* src, const int* dst,
                                float* alpha, float* amax, int E, int H, int C) {
  int idx = blockIdx.x * blockDim.x + threadIdx.x;
  if (idx >= E * H) return;
  int e = idx / H, h = idx - e * H;
  int s = src[e], d = dst[e];
  int HC = H * C;
  float eaf[7];
  const float* eap = ea + (size_t)e * 7;
#pragma unroll
  for (int f = 0; f < 7; ++f) eaf[f] = eap[f];
  const float* qp = Q + (size_t)d * HC + h * C;
  const float* kp = Km + (size_t)s * HC + h * C;
  const float* wep = we + h * C;
  const float* bep = be + h * C;
  float acc = 0.f;
  for (int c = 0; c < C; ++c) {
    float ee = bep[c];
#pragma unroll
    for (int f = 0; f < 7; ++f) ee = fmaf(eaf[f], wep[(size_t)f * HC + c], ee);
    acc = fmaf(qp[c], kp[c] + ee, acc);
  }
  acc *= rsqrtf((float)C);
  alpha[idx] = acc;
  atomicMaxF(&amax[d * H + h], acc);
}

__global__ void tr_scatter_kernel(const float* V, const float* ea, const float* we,
                                  const float* be, const float* alpha, const float* asum,
                                  const int* src, const int* dst, float* out,
                                  int E, int H, int C) {
  int e = blockIdx.x;
  int t = blockIdx.y * blockDim.x + threadIdx.x;
  int HC = H * C;
  if (t >= HC) return;
  int s = src[e], d = dst[e];
  int h = t / C;
  const float* eap = ea + (size_t)e * 7;
  float ee = be[t];
#pragma unroll
  for (int f = 0; f < 7; ++f) ee = fmaf(eap[f], we[(size_t)f * HC + t], ee);
  float a = alpha[(size_t)e * H + h] / (asum[d * H + h] + 1e-16f);
  atomicAdd(&out[(size_t)d * HC + t], (V[(size_t)s * HC + t] + ee) * a);
}

// ---------------------------------------------------------------------------
// GIN / GCN kernels
// ---------------------------------------------------------------------------
__global__ void gin_combine_kernel(const float* X, const float* agg, const float* eps,
                                   float* out, size_t n) {
  size_t i = blockIdx.x * (size_t)blockDim.x + threadIdx.x;
  if (i < n) out[i] = (1.0f + eps[0]) * X[i] + agg[i];
}

__global__ void gcn_dinv_kernel(float* deg, int n) {
  int i = blockIdx.x * blockDim.x + threadIdx.x;
  if (i < n) deg[i] = rsqrtf(deg[i] + 1.0f);   // +1 self loop; always > 0
}

__global__ void gcn_scatter_kernel(const float* XW, const float* dinv,
                                   const int* src, const int* dst, float* out,
                                   int E, int F) {
  long idx = blockIdx.x * (long)blockDim.x + threadIdx.x;
  if (idx >= (long)E * F) return;
  int e = (int)(idx / F), f = (int)(idx - (long)e * F);
  int s = src[e], d = dst[e];
  atomicAdd(&out[(size_t)d * F + f], XW[(size_t)s * F + f] * dinv[s] * dinv[d]);
}

__global__ void gcn_self_kernel(const float* XW, const float* dinv, float* out,
                                int n, int F) {
  long idx = blockIdx.x * (long)blockDim.x + threadIdx.x;
  if (idx >= (long)n * F) return;
  int i = (int)(idx / F), f = (int)(idx - (long)i * F);
  atomicAdd(&out[(size_t)i * F + f], XW[(size_t)i * F + f] * dinv[i] * dinv[i]);
}

// ---------------------------------------------------------------------------
// Pool3 (mean | max | sum) per graph
// ---------------------------------------------------------------------------
__global__ void pool_init_kernel(float* pool, int ld, int F) {
  int idx = blockIdx.x * blockDim.x + threadIdx.x;
  if (idx >= 64 * 3 * F) return;
  int g = idx / (3 * F), j = idx - g * 3 * F;
  pool[(size_t)g * ld + j] = (j >= F && j < 2 * F) ? -INFINITY : 0.0f;
}

__global__ void pool_scatter_kernel(const float* X, const int* batch, float* pool,
                                    float* cntg, int ld, int n, int F) {
  long idx = blockIdx.x * (long)blockDim.x + threadIdx.x;
  if (idx >= (long)n * F) return;
  int i = (int)(idx / F), f = (int)(idx - (long)i * F);
  int g = batch[i];
  float v = X[(size_t)i * F + f];
  atomicAdd(&pool[(size_t)g * ld + 2 * F + f], v);
  atomicMaxF(&pool[(size_t)g * ld + F + f], v);
  if (f == 0) atomicAdd(&cntg[g], 1.0f);
}

__global__ void pool_final_kernel(float* pool, const float* cntg, int ld, int F) {
  int idx = blockIdx.x * blockDim.x + threadIdx.x;
  if (idx >= 64 * F) return;
  int g = idx / F, f = idx - g * F;
  float c = fmaxf(cntg[g], 1.0f);
  pool[(size_t)g * ld + f] = pool[(size_t)g * ld + 2 * F + f] / c;
  float m = pool[(size_t)g * ld + F + f];
  if (!isfinite(m)) pool[(size_t)g * ld + F + f] = 0.0f;
}

// ===========================================================================
// Host orchestration
// ===========================================================================
struct GatP  { const float *wl,*bl,*wr,*br,*we,*att,*bias; };
struct TrP   { const float *wq,*bq,*wk,*bk,*wv,*bv,*we,*be,*ws,*bs; };
struct GinP  { const float *w1,*b1,*w2,*b2,*eps; };
struct GcnP  { const float *w,*b; };
struct SageP { const float *wl,*bl,*wr; };
struct HeadP { const float *w[4],*b[4],*g[4],*beta[4]; const float *wo,*bo; };

struct Ctx {
  hipStream_t st;
  const int *src, *dst, *batch;
  const float *edge_attr;
  int n, E;
  unsigned short *WPACK, *APACK;
  float *HA,*HB,*P0,*P1,*P2;
  float *EALPHA,*AMAX,*ASUM,*LOOPATTR,*CNTN,*POOL,*CNTG,*T0,*T1,*TOUT;
};

static inline dim3 g1d(long tot) { return dim3((unsigned)((tot + 255) / 256)); }

static void gemm(Ctx& c, const float* A, const float* W, const float* bias,
                 float* C, int M, int Nn, int K, int act, bool accum,
                 bool packA = true) {
  int Kpad = (K + 31) & ~31;
  if (packA)
    apack_kernel<<<g1d((long)M * Kpad), 256, 0, c.st>>>(A, c.APACK, K, Kpad, M);
  prepack_kernel<<<g1d((long)Nn * Kpad), 256, 0, c.st>>>(W, c.WPACK, K, Nn, Kpad);
  dim3 grid((Nn + 63) / 64, (M + 63) / 64), blk(128);
  if (!accum) {
    if (act == 0)      gemm_wmma_kernel<0,0><<<grid, blk, 0, c.st>>>(c.APACK, c.WPACK, bias, C, M, Nn, Kpad);
    else if (act == 1) gemm_wmma_kernel<1,0><<<grid, blk, 0, c.st>>>(c.APACK, c.WPACK, bias, C, M, Nn, Kpad);
    else               gemm_wmma_kernel<2,0><<<grid, blk, 0, c.st>>>(c.APACK, c.WPACK, bias, C, M, Nn, Kpad);
  } else {
    if (act == 0)      gemm_wmma_kernel<0,1><<<grid, blk, 0, c.st>>>(c.APACK, c.WPACK, bias, C, M, Nn, Kpad);
    else if (act == 1) gemm_wmma_kernel<1,1><<<grid, blk, 0, c.st>>>(c.APACK, c.WPACK, bias, C, M, Nn, Kpad);
    else               gemm_wmma_kernel<2,1><<<grid, blk, 0, c.st>>>(c.APACK, c.WPACK, bias, C, M, Nn, Kpad);
  }
}

static void run_gatv2(Ctx& c, const float* X, int IN, int H, int C, bool concat,
                      const GatP& p, float* out) {
  int n = c.n, E = c.E, HC = H * C;
  gemm(c, X, p.wl, p.bl, c.P0, n, HC, IN, 0, false, true);    // xl (packs X)
  gemm(c, X, p.wr, p.br, c.P1, n, HC, IN, 0, false, false);   // xr (reuses pack)
  fill_kernel<<<g1d((long)n * H), 256, 0, c.st>>>(c.AMAX, -INFINITY, (size_t)n * H);
  hipMemsetAsync(c.ASUM, 0, (size_t)n * H * 4, c.st);
  long tot = (long)(E + n) * H;
  gat_alpha_kernel<<<g1d(tot), 256, 0, c.st>>>(c.P0, c.P1, c.edge_attr, c.LOOPATTR,
                                               p.we, p.att, c.src, c.dst,
                                               c.EALPHA, c.AMAX, E, n, H, C);
  seg_exp_kernel<<<g1d(tot), 256, 0, c.st>>>(c.EALPHA, c.AMAX, c.ASUM, c.dst, E, n, H);
  float* sc = concat ? out : c.P2;
  hipMemsetAsync(sc, 0, (size_t)n * HC * 4, c.st);
  dim3 gs(E + n, (HC + 255) / 256);
  gat_scatter_kernel<<<gs, 256, 0, c.st>>>(c.P0, c.EALPHA, c.ASUM, c.src, c.dst,
                                           sc, E, n, H, C);
  if (concat)
    bias_act_kernel<1><<<g1d((long)n * HC), 256, 0, c.st>>>(out, p.bias, out, (size_t)n, HC);
  else
    head_mean_kernel<1><<<g1d((long)n * C), 256, 0, c.st>>>(c.P2, p.bias, out, n, H, C);
}

static void run_tr(Ctx& c, const float* X, int IN, int H, int C, bool concat,
                   const TrP& p, float* out) {
  int n = c.n, E = c.E, HC = H * C;
  gemm(c, X, p.wq, p.bq, c.P0, n, HC, IN, 0, false, true);    // q (packs X)
  gemm(c, X, p.wk, p.bk, c.P1, n, HC, IN, 0, false, false);   // k
  gemm(c, X, p.wv, p.bv, c.P2, n, HC, IN, 0, false, false);   // v
  fill_kernel<<<g1d((long)n * H), 256, 0, c.st>>>(c.AMAX, -INFINITY, (size_t)n * H);
  hipMemsetAsync(c.ASUM, 0, (size_t)n * H * 4, c.st);
  long tot = (long)E * H;
  tr_alpha_kernel<<<g1d(tot), 256, 0, c.st>>>(c.P0, c.P1, c.edge_attr, p.we, p.be,
                                              c.src, c.dst, c.EALPHA, c.AMAX, E, H, C);
  seg_exp_kernel<<<g1d(tot), 256, 0, c.st>>>(c.EALPHA, c.AMAX, c.ASUM, c.dst, E, 0, H);
  dim3 gs(E, (HC + 255) / 256);
  if (concat) {
    hipMemsetAsync(out, 0, (size_t)n * HC * 4, c.st);
    tr_scatter_kernel<<<gs, 256, 0, c.st>>>(c.P2, c.edge_attr, p.we, p.be, c.EALPHA,
                                            c.ASUM, c.src, c.dst, out, E, H, C);
    gemm(c, X, p.ws, p.bs, out, n, HC, IN, 1, true, false);   // reuse pack of X
  } else {
    hipMemsetAsync(c.P0, 0, (size_t)n * HC * 4, c.st); // q no longer needed
    tr_scatter_kernel<<<gs, 256, 0, c.st>>>(c.P2, c.edge_attr, p.we, p.be, c.EALPHA,
                                            c.ASUM, c.src, c.dst, c.P0, E, H, C);
    head_mean_kernel<0><<<g1d((long)n * C), 256, 0, c.st>>>(c.P0, nullptr, out, n, H, C);
    gemm(c, X, p.ws, p.bs, out, n, C, IN, 1, true, false);
  }
}

static void run_gin(Ctx& c, const float* X, int fin, int fh, int fout,
                    const GinP& p, float* out) {
  int n = c.n, E = c.E;
  hipMemsetAsync(c.P0, 0, (size_t)n * fin * 4, c.st);
  scatter_add_feat_kernel<<<g1d((long)E * fin), 256, 0, c.st>>>(X, c.src, c.dst, c.P0, E, fin);
  gin_combine_kernel<<<g1d((long)n * fin), 256, 0, c.st>>>(X, c.P0, p.eps, c.P1, (size_t)n * fin);
  gemm(c, c.P1, p.w1, p.b1, c.P2, n, fh, fin, 2, false);
  gemm(c, c.P2, p.w2, p.b2, out, n, fout, fh, 2, false);  // outer relu folded
}

static void run_gcn(Ctx& c, const float* X, int fin, int fout, const GcnP& p, float* out) {
  int n = c.n, E = c.E;
  hipMemsetAsync(c.CNTN, 0, (size_t)n * 4, c.st);
  count_dst_kernel<<<g1d(E), 256, 0, c.st>>>(c.dst, c.CNTN, E);
  gcn_dinv_kernel<<<g1d(n), 256, 0, c.st>>>(c.CNTN, n);
  gemm(c, X, p.w, nullptr, c.P0, n, fout, fin, 0, false);
  hipMemsetAsync(c.P1, 0, (size_t)n * fout * 4, c.st);
  gcn_scatter_kernel<<<g1d((long)E * fout), 256, 0, c.st>>>(c.P0, c.CNTN, c.src, c.dst, c.P1, E, fout);
  gcn_self_kernel<<<g1d((long)n * fout), 256, 0, c.st>>>(c.P0, c.CNTN, c.P1, n, fout);
  bias_act_kernel<2><<<g1d((long)n * fout), 256, 0, c.st>>>(c.P1, p.b, out, (size_t)n, fout);
}

static void run_sage(Ctx& c, const float* X, int fin, int fout, const SageP& p,
                     bool mean, float* out) {
  int n = c.n, E = c.E;
  if (mean) {
    hipMemsetAsync(c.P0, 0, (size_t)n * fin * 4, c.st);
    hipMemsetAsync(c.CNTN, 0, (size_t)n * 4, c.st);
    scatter_add_feat_kernel<<<g1d((long)E * fin), 256, 0, c.st>>>(X, c.src, c.dst, c.P0, E, fin);
    count_dst_kernel<<<g1d(E), 256, 0, c.st>>>(c.dst, c.CNTN, E);
    div_cnt_kernel<<<g1d((long)n * fin), 256, 0, c.st>>>(c.P0, c.CNTN, n, fin);
  } else {
    fill_kernel<<<g1d((long)n * fin), 256, 0, c.st>>>(c.P0, -INFINITY, (size_t)n * fin);
    scatter_max_feat_kernel<<<g1d((long)E * fin), 256, 0, c.st>>>(X, c.src, c.dst, c.P0, E, fin);
    fixinf_kernel<<<g1d((long)n * fin), 256, 0, c.st>>>(c.P0, (size_t)n * fin);
  }
  gemm(c, c.P0, p.wl, p.bl, out, n, fout, fin, 0, false);
  gemm(c, X, p.wr, nullptr, out, n, fout, fin, 2, true);  // outer relu folded
}

static void run_pool(Ctx& c, const float* X, int F, float* pool_out, int ld) {
  hipMemsetAsync(c.CNTG, 0, 64 * 4, c.st);
  pool_init_kernel<<<g1d(64L * 3 * F), 256, 0, c.st>>>(pool_out, ld, F);
  pool_scatter_kernel<<<g1d((long)c.n * F), 256, 0, c.st>>>(X, c.batch, pool_out, c.CNTG, ld, c.n, F);
  pool_final_kernel<<<g1d(64L * F), 256, 0, c.st>>>(pool_out, c.CNTG, ld, F);
}

static void run_head(Ctx& c, const HeadP& h, const int* dims, const float* in,
                     float coef, float* dout) {
  const float* cur = in;
  float* bufs[2] = {c.T0, c.T1};
  int bi = 0;
  for (int i = 0; i < 4; ++i) {
    float* t = bufs[bi]; bi ^= 1;
    gemm(c, cur, h.w[i], h.b[i], t, 64, dims[i + 1], dims[i], 1, false);
    layernorm_kernel<<<1, 64, 0, c.st>>>(t, h.g[i], h.beta[i], 64, dims[i + 1]);
    cur = t;
  }
  gemm(c, cur, h.wo, h.bo, c.TOUT, 64, 9, dims[4], 0, false);
  axpy_kernel<<<g1d(576), 256, 0, c.st>>>(dout, c.TOUT, coef, 576);
}

// ===========================================================================
extern "C" void kernel_launch(void* const* d_in, const int* in_sizes, int n_in,
                              void* d_out, int out_size, void* d_ws, size_t ws_size,
                              hipStream_t stream) {
  (void)n_in; (void)out_size; (void)ws_size;

  int n = in_sizes[0] / 14;
  int E = in_sizes[1] / 7;

  // ---- unpack parameters in setup_inputs() insertion order ----
  int ip = 0;
  auto nf = [&]() { return (const float*)d_in[ip++]; };
  const float* x = nf();
  const float* edge_attr = nf();

  GatP gat_c[4]; HeadP gat_h;
  for (int i = 0; i < 4; ++i)
    gat_c[i] = {nf(), nf(), nf(), nf(), nf(), nf(), nf()};
  for (int i = 0; i < 4; ++i) { gat_h.w[i]=nf(); gat_h.b[i]=nf(); gat_h.g[i]=nf(); gat_h.beta[i]=nf(); }
  gat_h.wo = nf(); gat_h.bo = nf();

  TrP tr_c[4]; HeadP tr_h;
  for (int i = 0; i < 4; ++i)
    tr_c[i] = {nf(), nf(), nf(), nf(), nf(), nf(), nf(), nf(), nf(), nf()};
  for (int i = 0; i < 4; ++i) { tr_h.w[i]=nf(); tr_h.b[i]=nf(); tr_h.g[i]=nf(); tr_h.beta[i]=nf(); }
  tr_h.wo = nf(); tr_h.bo = nf();

  GinP gin_c[3]; HeadP gin_h;
  for (int i = 0; i < 3; ++i) gin_c[i] = {nf(), nf(), nf(), nf(), nf()};
  for (int i = 0; i < 4; ++i) { gin_h.w[i]=nf(); gin_h.b[i]=nf(); gin_h.g[i]=nf(); gin_h.beta[i]=nf(); }
  gin_h.wo = nf(); gin_h.bo = nf();

  GatP hyb_g[2]; GcnP hyb_c[2]; HeadP hyb_h;
  for (int i = 0; i < 2; ++i) hyb_g[i] = {nf(), nf(), nf(), nf(), nf(), nf(), nf()};
  for (int i = 0; i < 2; ++i) hyb_c[i] = {nf(), nf()};
  for (int i = 0; i < 4; ++i) { hyb_h.w[i]=nf(); hyb_h.b[i]=nf(); hyb_h.g[i]=nf(); hyb_h.beta[i]=nf(); }
  hyb_h.wo = nf(); hyb_h.bo = nf();

  SageP sage_c[3]; HeadP sage_h;
  for (int i = 0; i < 3; ++i) sage_c[i] = {nf(), nf(), nf()};
  for (int i = 0; i < 4; ++i) { sage_h.w[i]=nf(); sage_h.b[i]=nf(); sage_h.g[i]=nf(); sage_h.beta[i]=nf(); }
  sage_h.wo = nf(); sage_h.bo = nf();

  const int* edge_index = (const int*)d_in[ip++];
  const int* batch      = (const int*)d_in[ip++];

  // ---- workspace layout ----
  char* base = (char*)d_ws;
  size_t off = 0;
  auto alloc = [&](size_t bytes) { void* p = base + off; off += (bytes + 255) & ~(size_t)255; return p; };
  size_t nf768 = (size_t)n * 768 * sizeof(float);

  Ctx c;
  c.st = stream;
  c.src = edge_index; c.dst = edge_index + E; c.batch = batch;
  c.edge_attr = edge_attr; c.n = n; c.E = E;
  c.WPACK    = (unsigned short*)alloc(768 * 768 * sizeof(unsigned short));
  c.APACK    = (unsigned short*)alloc((size_t)n * 768 * sizeof(unsigned short));
  c.HA       = (float*)alloc(nf768);
  c.HB       = (float*)alloc(nf768);
  c.P0       = (float*)alloc(nf768);
  c.P1       = (float*)alloc(nf768);
  c.P2       = (float*)alloc(nf768);
  c.EALPHA   = (float*)alloc((size_t)(E + n) * 6 * sizeof(float));
  c.AMAX     = (float*)alloc((size_t)n * 6 * sizeof(float));
  c.ASUM     = (float*)alloc((size_t)n * 6 * sizeof(float));
  c.LOOPATTR = (float*)alloc((size_t)n * 7 * sizeof(float));
  c.CNTN     = (float*)alloc((size_t)n * sizeof(float));
  c.POOL     = (float*)alloc(64 * 384 * sizeof(float));
  c.CNTG     = (float*)alloc(64 * sizeof(float));
  c.T0       = (float*)alloc(64 * 384 * sizeof(float));
  c.T1       = (float*)alloc(64 * 384 * sizeof(float));
  c.TOUT     = (float*)alloc(64 * 16 * sizeof(float));

  hipMemsetAsync(d_out, 0, 64 * 9 * sizeof(float), stream);
  float* dout = (float*)d_out;

  // self-loop edge_attr mean (shared by every GATv2 call)
  hipMemsetAsync(c.LOOPATTR, 0, (size_t)n * 7 * 4, stream);
  hipMemsetAsync(c.CNTN, 0, (size_t)n * 4, stream);
  scatter_add_feat_kernel<<<g1d((long)E * 7), 256, 0, stream>>>(edge_attr, c.src, c.dst, c.LOOPATTR, E, 7);
  count_dst_kernel<<<g1d(E), 256, 0, stream>>>(c.dst, c.CNTN, E);
  div_cnt_kernel<<<g1d((long)n * 7), 256, 0, stream>>>(c.LOOPATTR, c.CNTN, n, 7);

  // ---- Model 1: GATv2 ----
  run_gatv2(c, x,    14,  6, 128, true,  gat_c[0], c.HA);
  run_gatv2(c, c.HA, 768, 6, 128, true,  gat_c[1], c.HB);
  run_gatv2(c, c.HB, 768, 4, 64,  true,  gat_c[2], c.HA);
  run_gatv2(c, c.HA, 256, 2, 64,  false, gat_c[3], c.HB);
  run_pool(c, c.HB, 64, c.POOL, 192);
  { int dims[6] = {192, 384, 256, 128, 64, 9};
    run_head(c, gat_h, dims, c.POOL, 0.2f, dout); }

  // ---- Model 2: TransformerConv ----
  run_tr(c, x,    14,  6, 128, true,  tr_c[0], c.HA);
  run_tr(c, c.HA, 768, 6, 128, true,  tr_c[1], c.HB);
  run_tr(c, c.HB, 768, 4, 64,  true,  tr_c[2], c.HA);
  run_tr(c, c.HA, 256, 2, 64,  false, tr_c[3], c.HB);
  run_pool(c, c.HB, 64, c.POOL, 192);
  { int dims[6] = {192, 320, 192, 96, 48, 9};
    run_head(c, tr_h, dims, c.POOL, 0.35f, dout); }

  // ---- Model 3: GIN (weight 0 but still computed) ----
  run_gin(c, x,    14,  128, 128, gin_c[0], c.HA);
  run_gin(c, c.HA, 128, 128, 128, gin_c[1], c.HB);
  run_gin(c, c.HB, 128, 64,  64,  gin_c[2], c.HA);
  run_pool(c, c.HA, 64, c.POOL, 192);
  { int dims[6] = {192, 256, 192, 128, 64, 9};
    run_head(c, gin_h, dims, c.POOL, 0.0f, dout); }

  // ---- Model 4: Hybrid GATv2 + GCN ----
  run_gatv2(c, x,    14,  4, 64, true,  hyb_g[0], c.HA);
  run_gatv2(c, c.HA, 256, 2, 64, false, hyb_g[1], c.HB);
  run_pool(c, c.HB, 64, c.POOL, 384);
  run_gcn(c, x,    14, 64, hyb_c[0], c.HA);
  run_gcn(c, c.HA, 64, 64, hyb_c[1], c.HB);
  run_pool(c, c.HB, 64, c.POOL + 192, 384);
  { int dims[6] = {384, 320, 256, 128, 64, 9};
    run_head(c, hyb_h, dims, c.POOL, 0.35f, dout); }

  // ---- Model 5: GraphSAGE ----
  run_sage(c, x,    14,  128, sage_c[0], true,  c.HA);
  run_sage(c, c.HA, 128, 128, sage_c[1], false, c.HB);
  run_sage(c, c.HB, 128, 64,  64 ? sage_c[2] : sage_c[2], true,  c.HA);
  run_pool(c, c.HA, 64, c.POOL, 192);
  { int dims[6] = {192, 256, 192, 128, 64, 9};
    run_head(c, sage_h, dims, c.POOL, 0.1f, dout); }
}
